// NerModelWord_49082886258821
// MI455X (gfx1250) — matrix operand order repaired and checked by reading, not compiled, not run on hardware
//
#include <hip/hip_runtime.h>

// ---------------------------------------------------------------------------
// BiLSTM-CRF NER for MI455X (gfx1250), wave32 + WMMA bf16 tensor path.
// Round 4: replace libm tanhf in the recurrent gate math with the
// exp/rcp-based identity tanh(x)=2*sigmoid(2x)-1 (fast v_exp/v_rcp path) —
// the scalar tail was rivaling the per-step WMMA work on the critical path.
// ---------------------------------------------------------------------------

typedef __attribute__((ext_vector_type(16))) __bf16 v16bf;
typedef __attribute__((ext_vector_type(8)))  __bf16 v8bf;
typedef __attribute__((ext_vector_type(8)))  float  v8f;

#define Bn   64
#define Tn   256
#define En   128
#define Hn   256
#define Ln   9
#define G4H  1024        // 4*H
#define Kc   384         // E + H (fused [x_t | h] GEMM K)

__device__ __forceinline__ float fast_sigmoid(float x) {
  return __fdividef(1.0f, 1.0f + __expf(-x));
}
__device__ __forceinline__ float fast_tanh(float x) {
  return __fdividef(2.0f, 1.0f + __expf(-2.0f * x)) - 1.0f;
}

// ---------------------------------------------------------------------------
// Pack Wi (E x 4H) and Wh (H x 4H) into bf16 Wc[n][k], n in [0,4H), k in [0,384).
// B-fragment loads become one contiguous 32-byte load per lane.
// ---------------------------------------------------------------------------
__global__ void k_pack(const float* __restrict__ Wi, const float* __restrict__ Wh,
                       __bf16* __restrict__ Wc) {
  int idx = blockIdx.x * blockDim.x + threadIdx.x;       // idx = n*Kc + k
  if (idx >= G4H * Kc) return;
  int n = idx / Kc, k = idx % Kc;
  float v = (k < En) ? Wi[(size_t)k * G4H + n] : Wh[(size_t)(k - En) * G4H + n];
  Wc[idx] = (__bf16)v;
}

// ---------------------------------------------------------------------------
// x[b,t,:] = char_emb[char] + word_emb[word] @ W_word  -> bf16 (WMMA A operand)
// One block per (b,t) row; word row staged in LDS.
// ---------------------------------------------------------------------------
__global__ __launch_bounds__(128) void k_embed(
    const int* __restrict__ ch, const int* __restrict__ wd,
    const float* __restrict__ cemb, const float* __restrict__ wemb,
    const float* __restrict__ Wword, __bf16* __restrict__ xbf) {
  const int row = blockIdx.x;          // b*T + t
  const int e = threadIdx.x;
  __shared__ float wrow[100];
  const int w = wd[row];
  if (e < 100) wrow[e] = wemb[(size_t)w * 100 + e];
  __syncthreads();
  float s = cemb[(size_t)ch[row] * En + e];
#pragma unroll 4
  for (int k = 0; k < 100; ++k) s += wrow[k] * Wword[k * En + e];
  xbf[(size_t)row * En + e] = (__bf16)s;
}

// ---------------------------------------------------------------------------
// Fused LSTM recurrence. grid = 2 blocks (dir 0 = fwd, dir 1 = bwd),
// 512 threads = 16 wave32s on one WGP. Per step:
//   z(64x1024) = [x_t | h](64x384) @ Wc(384x1024) + bias   (WMMA bf16)
//   i,f,g,o gates wave-local; c in VGPRs; h -> LDS(bf16) + global(f32).
// Wave w owns hidden columns [16w,16w+16) for all 4 gates and all 4 M-tiles.
// ---------------------------------------------------------------------------
__global__ __launch_bounds__(512) void k_lstm(
    const __bf16* __restrict__ xbf,     // [B][T][E]
    const __bf16* __restrict__ WcF,     // [4H][Kc]
    const __bf16* __restrict__ WcB,
    const float* __restrict__ biasF,    // [4H]
    const float* __restrict__ biasB,
    float* __restrict__ hsF,            // [T][B][H]
    float* __restrict__ hsB) {
  const int dir = blockIdx.x;
  const __bf16* __restrict__ Wc  = dir ? WcB : WcF;
  const float*  __restrict__ bias = dir ? biasB : biasF;
  float* __restrict__ hs = dir ? hsB : hsF;

  const int tid  = threadIdx.x;
  const int lane = tid & 31;
  const int wv   = tid >> 5;        // wave 0..15
  const int hl   = lane >> 4;       // lane half (selects K sub-block)
  const int col  = lane & 15;       // M row (A) / N col (B,C)
  const int jb   = wv * 16;         // hidden column block for this wave

  __shared__ __bf16 h_lds[Bn * Hn]; // 32 KB

  for (int i = tid; i < Bn * Hn; i += 512) h_lds[i] = (__bf16)0.0f;
  __syncthreads();

  float bg[4];
#pragma unroll
  for (int g = 0; g < 4; ++g) bg[g] = bias[g * Hn + jb + col];

  // Per-gate weight base pointers for this wave's column block.
  const __bf16* wptr[4];
#pragma unroll
  for (int g = 0; g < 4; ++g)
    wptr[g] = Wc + (size_t)(g * Hn + jb + col) * Kc + hl * 16;

  v8f c[4];
#pragma unroll
  for (int m = 0; m < 4; ++m)
#pragma unroll
    for (int e = 0; e < 8; ++e) c[m][e] = 0.0f;

#pragma unroll 1
  for (int s = 0; s < Tn; ++s) {
    const int t = dir ? (Tn - 1 - s) : s;

    v8f acc[4][4];
#pragma unroll
    for (int m = 0; m < 4; ++m)
#pragma unroll
      for (int g = 0; g < 4; ++g)
#pragma unroll
        for (int e = 0; e < 8; ++e) acc[m][g][e] = bg[g];

    // Phase 1: K = x_t part (4 k-tiles, A from global x). Branch-free.
#pragma unroll 1
    for (int kt = 0; kt < 4; ++kt) {
      v16bf bf[4];
#pragma unroll
      for (int g = 0; g < 4; ++g)
        bf[g] = *(const v16bf*)(wptr[g] + kt * 32);
#pragma unroll
      for (int g = 0; g < 4; ++g)            // speculative, always issued
        __builtin_prefetch(wptr[g] + (kt + 1) * 32, 0, 3);
#pragma unroll
      for (int m = 0; m < 4; ++m) {
        const int row = m * 16 + col;
        const __bf16* px = xbf + ((size_t)row * Tn + t) * En + kt * 32 + hl * 16;
        v8bf lo = *(const v8bf*)px;
        v8bf hi = *(const v8bf*)(px + 8);
        v16bf a;
#pragma unroll
        for (int e = 0; e < 8; ++e) { a[e] = lo[e]; a[e + 8] = hi[e]; }
#pragma unroll
        for (int g = 0; g < 4; ++g)
          acc[m][g] = __builtin_amdgcn_wmma_f32_16x16x32_bf16(
              false, a, false, bf[g], (short)0, acc[m][g], false, false);
      }
    }

    // Phase 2: K = h part (8 k-tiles, A from LDS). Branch-free.
#pragma unroll 1
    for (int kt = 0; kt < 8; ++kt) {
      v16bf bf[4];
#pragma unroll
      for (int g = 0; g < 4; ++g)
        bf[g] = *(const v16bf*)(wptr[g] + (4 + kt) * 32);
#pragma unroll
      for (int g = 0; g < 4; ++g)            // speculative, always issued
        __builtin_prefetch(wptr[g] + (5 + kt) * 32, 0, 3);
#pragma unroll
      for (int m = 0; m < 4; ++m) {
        const int row = m * 16 + col;
        const __bf16* ph = h_lds + row * Hn + kt * 32 + hl * 16;
        v8bf lo = *(const v8bf*)ph;
        v8bf hi = *(const v8bf*)(ph + 8);
        v16bf a;
#pragma unroll
        for (int e = 0; e < 8; ++e) { a[e] = lo[e]; a[e + 8] = hi[e]; }
#pragma unroll
        for (int g = 0; g < 4; ++g)
          acc[m][g] = __builtin_amdgcn_wmma_f32_16x16x32_bf16(
              false, a, false, bf[g], (short)0, acc[m][g], false, false);
      }
    }

    // Gate nonlinearities, wave-local cell update (fast exp/rcp path).
    v8f hv[4];
#pragma unroll
    for (int m = 0; m < 4; ++m) {
#pragma unroll
      for (int e = 0; e < 8; ++e) {
        float zi = acc[m][0][e], zf = acc[m][1][e];
        float zg = acc[m][2][e], zo = acc[m][3][e];
        float si = fast_sigmoid(zi);
        float sf = fast_sigmoid(zf);
        float so = fast_sigmoid(zo);
        float cn = sf * c[m][e] + si * fast_tanh(zg);
        c[m][e] = cn;
        hv[m][e] = so * fast_tanh(cn);
      }
    }

    __syncthreads();   // all waves finished reading previous h
#pragma unroll
    for (int m = 0; m < 4; ++m) {
#pragma unroll
      for (int r = 0; r < 8; ++r) {
        const int row = m * 16 + hl * 8 + r;   // C layout: VGPR r, lane-half -> M
        const int j = jb + col;
        h_lds[row * Hn + j] = (__bf16)hv[m][r];
        hs[(size_t)t * (Bn * Hn) + (size_t)row * Hn + j] = hv[m][r];
      }
    }
    __syncthreads();   // new h visible for next step
  }
}

// ---------------------------------------------------------------------------
// logits = [hf|hb] @ W_out + b_out.  N=9 -> tiny GEMM, bandwidth-trivial:
// one thread per output element, K=512 dot product out of L2.
// ---------------------------------------------------------------------------
__global__ void k_logits(const float* __restrict__ hsF, const float* __restrict__ hsB,
                         const float* __restrict__ Wout, const float* __restrict__ bout,
                         float* __restrict__ out) {
  int idx = blockIdx.x * blockDim.x + threadIdx.x;
  if (idx >= Bn * Tn * Ln) return;
  int r = idx / Ln, l = idx % Ln;
  int b = r / Tn, t = r % Tn;
  const float* hf = hsF + (size_t)t * (Bn * Hn) + (size_t)b * Hn;
  const float* hb = hsB + (size_t)t * (Bn * Hn) + (size_t)b * Hn;
  float s = bout[l];
#pragma unroll 4
  for (int j = 0; j < Hn; ++j) {
    s += hf[j] * Wout[j * Ln + l];
    s += hb[j] * Wout[(Hn + j) * Ln + l];
  }
  out[idx] = s;
}

// ---------------------------------------------------------------------------
__global__ void k_lens(const int* __restrict__ ch, float* __restrict__ out_lens,
                       int* __restrict__ lens_i) {
  int b = threadIdx.x;
  if (b >= Bn) return;
  int n = 0;
  for (int t = 0; t < Tn; ++t) n += (ch[b * Tn + t] != 0) ? 1 : 0;
  out_lens[b] = (float)n;
  lens_i[b] = n;
}

// ---------------------------------------------------------------------------
// ll = crf_score - crf_log_norm (forward algorithm, 9 states). One thread/batch.
// ---------------------------------------------------------------------------
__global__ void k_crf_ll(const float* __restrict__ logits, const int* __restrict__ labels,
                         const float* __restrict__ trans, const int* __restrict__ lens_i,
                         float* __restrict__ out_ll) {
  int b = blockIdx.x * blockDim.x + threadIdx.x;
  if (b >= Bn) return;
  const int len = lens_i[b];
  const float* lg = logits + (size_t)b * Tn * Ln;
  const int* lb = labels + b * Tn;

  float score = 0.0f;
  for (int t = 0; t < Tn; ++t) if (t < len) score += lg[t * Ln + lb[t]];
  for (int t = 1; t < Tn; ++t) if (t < len) score += trans[lb[t - 1] * Ln + lb[t]];

  float alpha[Ln];
#pragma unroll
  for (int l = 0; l < Ln; ++l) alpha[l] = lg[l];
  for (int t = 1; t < Tn; ++t) {
    if (t < len) {
      float na[Ln];
#pragma unroll
      for (int j = 0; j < Ln; ++j) {
        float mx = -1e30f;
#pragma unroll
        for (int i = 0; i < Ln; ++i) mx = fmaxf(mx, alpha[i] + trans[i * Ln + j]);
        float ssum = 0.0f;
#pragma unroll
        for (int i = 0; i < Ln; ++i) ssum += __expf(alpha[i] + trans[i * Ln + j] - mx);
        na[j] = mx + __logf(ssum) + lg[t * Ln + j];
      }
#pragma unroll
      for (int j = 0; j < Ln; ++j) alpha[j] = na[j];
    }
  }
  float mx = -1e30f;
#pragma unroll
  for (int l = 0; l < Ln; ++l) mx = fmaxf(mx, alpha[l]);
  float ssum = 0.0f;
#pragma unroll
  for (int l = 0; l < Ln; ++l) ssum += __expf(alpha[l] - mx);
  out_ll[b] = score - (mx + __logf(ssum));
}

// ---------------------------------------------------------------------------
// Viterbi decode: forward pass with backpointers in workspace, then backtrace.
// Matches reference masking (t >= len: alpha frozen, bp[j] = j). Tags as float.
// ---------------------------------------------------------------------------
__global__ void k_crf_decode(const float* __restrict__ logits, const float* __restrict__ trans,
                             const int* __restrict__ lens_i, int* __restrict__ bp,
                             float* __restrict__ out_tags) {
  int b = blockIdx.x * blockDim.x + threadIdx.x;
  if (b >= Bn) return;
  const int len = lens_i[b];
  const float* lg = logits + (size_t)b * Tn * Ln;

  float alpha[Ln];
#pragma unroll
  for (int l = 0; l < Ln; ++l) alpha[l] = lg[l];
  for (int t = 1; t < Tn; ++t) {
    int* bpt = bp + ((size_t)t * Bn + b) * Ln;
    if (t < len) {
      float na[Ln]; int nb[Ln];
#pragma unroll
      for (int j = 0; j < Ln; ++j) {
        float best = -1e30f; int arg = 0;
#pragma unroll
        for (int i = 0; i < Ln; ++i) {
          float v = alpha[i] + trans[i * Ln + j];
          if (v > best) { best = v; arg = i; }   // first max, like jnp.argmax
        }
        na[j] = best + lg[t * Ln + j]; nb[j] = arg;
      }
#pragma unroll
      for (int j = 0; j < Ln; ++j) { alpha[j] = na[j]; bpt[j] = nb[j]; }
    } else {
#pragma unroll
      for (int j = 0; j < Ln; ++j) bpt[j] = j;
    }
  }
  float best = -1e30f; int tag = 0;
#pragma unroll
  for (int l = 0; l < Ln; ++l) if (alpha[l] > best) { best = alpha[l]; tag = l; }
  out_tags[b * Tn + (Tn - 1)] = (float)tag;
  for (int t = Tn - 1; t >= 1; --t) {
    tag = bp[((size_t)t * Bn + b) * Ln + tag];
    out_tags[b * Tn + (t - 1)] = (float)tag;
  }
}

// ---------------------------------------------------------------------------
extern "C" void kernel_launch(void* const* d_in, const int* in_sizes, int n_in,
                              void* d_out, int out_size, void* d_ws, size_t ws_size,
                              hipStream_t stream) {
  (void)in_sizes; (void)n_in; (void)out_size; (void)ws_size;

  const int*   ch     = (const int*)  d_in[0];
  const int*   wd     = (const int*)  d_in[1];
  const int*   labels = (const int*)  d_in[2];
  const float* cemb   = (const float*)d_in[3];
  const float* wemb   = (const float*)d_in[4];
  const float* Wword  = (const float*)d_in[5];
  const float* WiF    = (const float*)d_in[6];
  const float* WhF    = (const float*)d_in[7];
  const float* bF     = (const float*)d_in[8];
  const float* WiB    = (const float*)d_in[9];
  const float* WhB    = (const float*)d_in[10];
  const float* bB     = (const float*)d_in[11];
  const float* Wout   = (const float*)d_in[12];
  const float* bout   = (const float*)d_in[13];
  const float* trans  = (const float*)d_in[14];

  char* ws = (char*)d_ws;
  __bf16* xbf  = (__bf16*)(ws + 0);           //  4,194,304 B : x bf16 [B][T][E]
  __bf16* WcF  = (__bf16*)(ws + 4194304);     //    786,432 B : packed fwd weights
  __bf16* WcB  = (__bf16*)(ws + 4980736);     //    786,432 B : packed bwd weights
  float*  hsF  = (float*) (ws + 5767168);     // 16,777,216 B : hf [T][B][H]
  float*  hsB  = (float*) (ws + 22544384);    // 16,777,216 B : hb [T][B][H]
  int*    bp   = (int*)   (ws + 39321600);    //    589,824 B : viterbi backptrs
  int*    lensI= (int*)   (ws + 39911424);    //        256 B : lens (int)

  float* out        = (float*)d_out;
  float* out_logits = out;                    // 147456
  float* out_lens   = out + 147456;           //     64
  float* out_ll     = out + 147520;           //     64
  float* out_tags   = out + 147584;           //  16384

  k_pack<<<(G4H * Kc + 255) / 256, 256, 0, stream>>>(WiF, WhF, WcF);
  k_pack<<<(G4H * Kc + 255) / 256, 256, 0, stream>>>(WiB, WhB, WcB);
  k_embed<<<Bn * Tn, 128, 0, stream>>>(ch, wd, cemb, wemb, Wword, xbf);
  k_lstm<<<2, 512, 0, stream>>>(xbf, WcF, WcB, bF, bB, hsF, hsB);
  k_logits<<<(Bn * Tn * Ln + 255) / 256, 256, 0, stream>>>(hsF, hsB, Wout, bout, out_logits);
  k_lens<<<1, 64, 0, stream>>>(ch, out_lens, lensI);
  k_crf_ll<<<1, 64, 0, stream>>>(out_logits, labels, trans, lensI, out_ll);
  k_crf_decode<<<1, 64, 0, stream>>>(out_logits, trans, lensI, bp, out_tags);
}
